// _BidirectionalMambaBlock_50809463111849
// MI455X (gfx1250) — compile-verified, matrix operand order checked by reading, not run on hardware
//
#include <hip/hip_runtime.h>
#include <hip/hip_bf16.h>

// ---------------------------------------------------------------------------
// Bidirectional Mamba block for MI455X (gfx1250), wave32.
// GEMMs: v_wmma_f32_16x16x32_bf16 with double-buffered LDS tiles staged via
// global_load_async_to_lds_b128 (ASYNCcnt) -- CDNA5 async copy path.
// ---------------------------------------------------------------------------

typedef __attribute__((ext_vector_type(16))) __bf16 v16bf;
typedef __attribute__((ext_vector_type(8)))  float  v8f;

#define B_SZ   2
#define L_SZ   2048
#define DM_SZ  1024
#define DI_SZ  2048
#define DS_SZ  16
#define DTR_SZ 64
#define M_SZ   (B_SZ * L_SZ)          // 4096 tokens

// ----------------------------- LayerNorm -----------------------------------
__global__ __launch_bounds__(256)
void mamba_layernorm_kernel(const float* __restrict__ x,
                            const float* __restrict__ gamma,
                            const float* __restrict__ beta,
                            __bf16* __restrict__ lnF,
                            __bf16* __restrict__ lnR) {
    const int row = blockIdx.x;                 // b*L + t
    const int b = row / L_SZ, t = row % L_SZ;
    const float* xr = x + (size_t)row * DM_SZ;
    __shared__ float red[256];

    float s = 0.f;
    for (int i = threadIdx.x; i < DM_SZ; i += 256) s += xr[i];
    red[threadIdx.x] = s; __syncthreads();
    for (int off = 128; off > 0; off >>= 1) {
        if (threadIdx.x < off) red[threadIdx.x] += red[threadIdx.x + off];
        __syncthreads();
    }
    const float mu = red[0] / (float)DM_SZ;
    __syncthreads();

    float v = 0.f;
    for (int i = threadIdx.x; i < DM_SZ; i += 256) { float d = xr[i] - mu; v += d * d; }
    red[threadIdx.x] = v; __syncthreads();
    for (int off = 128; off > 0; off >>= 1) {
        if (threadIdx.x < off) red[threadIdx.x] += red[threadIdx.x + off];
        __syncthreads();
    }
    const float rstd = rsqrtf(red[0] / (float)DM_SZ + 1e-5f);

    const size_t rf = (size_t)row * DM_SZ;
    const size_t rr = ((size_t)b * L_SZ + (size_t)(L_SZ - 1 - t)) * DM_SZ;
    for (int i = threadIdx.x; i < DM_SZ; i += 256) {
        float y = (xr[i] - mu) * rstd * gamma[i] + beta[i];
        lnF[rf + i] = (__bf16)y;
        lnR[rr + i] = (__bf16)y;
    }
}

// --------------------------- fp32 -> bf16 cast ------------------------------
__global__ void mamba_cvt_bf16_kernel(const float* __restrict__ src,
                                      __bf16* __restrict__ dst, size_t n) {
    size_t i = (size_t)blockIdx.x * 256 + threadIdx.x;
    if (i < n) dst[i] = (__bf16)src[i];
}

__global__ void mamba_slice_bf16_kernel(const float* __restrict__ src,
                                        __bf16* __restrict__ dst,
                                        int srcStride, int cols, size_t total) {
    size_t i = (size_t)blockIdx.x * 256 + threadIdx.x;
    if (i >= total) return;
    size_t r = i / cols; int c = (int)(i % cols);
    dst[i] = (__bf16)src[r * (size_t)srcStride + c];
}

// ------------------------------ WMMA GEMM -----------------------------------
// C[M,N] (fp32, row-major) = A[M,K] (bf16, row-major) * W[N,K]^T (bf16).
// Block tile 128x128, K-step 32, 8 waves; each wave owns a 32x64 patch made
// of 2x4 v_wmma_f32_16x16x32_bf16 accumulators. Full-N tiles use async
// global->LDS copies (ASYNCcnt) with double buffering; ragged-N tiles fall
// back to guarded synchronous staging.
#define GBLK  128
#define GBK   32
#define GLSTR 40   // LDS row stride in bf16: 80B rows -> 16B aligned, conflict-free

__device__ __forceinline__ void mamba_async_b128(unsigned lds_off, const void* gptr) {
    asm volatile("global_load_async_to_lds_b128 %0, %1, off"
                 :: "v"(lds_off), "v"(gptr)
                 : "memory");
}
__device__ __forceinline__ void mamba_wait_async0() {
    asm volatile("s_wait_asynccnt 0" ::: "memory");
}

__global__ __launch_bounds__(256)
void mamba_gemm_bf16_kernel(const __bf16* __restrict__ A,
                            const __bf16* __restrict__ W,
                            float* __restrict__ C,
                            int M, int N, int K) {
    __shared__ __align__(16) __bf16 As[2][GBLK * GLSTR];
    __shared__ __align__(16) __bf16 Ws[2][GBLK * GLSTR];

    const int tid   = threadIdx.x;
    const int lane  = tid & 31;
    const int wave  = tid >> 5;
    const int waveM = wave & 3;    // 4 wave-rows  * 32
    const int waveN = wave >> 2;   // 2 wave-cols  * 64
    const int bM    = blockIdx.x * GBLK;
    const int bN    = blockIdx.y * GBLK;
    const int ln15  = lane & 15;
    const int hf    = lane >> 4;

    const v8f vzero = {0.f, 0.f, 0.f, 0.f, 0.f, 0.f, 0.f, 0.f};
    v8f acc[2][4];
#pragma unroll
    for (int i = 0; i < 2; ++i)
#pragma unroll
        for (int j = 0; j < 4; ++j) acc[i][j] = vzero;

    // Fragment gather + 8 WMMAs from one LDS buffer (ISA 7.12.2 layouts:
    // dword v<4 : K = hf*8 + 2v ; v>=4 : K = 16 + hf*8 + 2(v-4)).
    auto compute = [&](const __bf16* as, const __bf16* ws) {
        union Frag { v16bf v; unsigned u[8]; };
        Frag fa[2], fb[4];
#pragma unroll
        for (int ms = 0; ms < 2; ++ms) {
            int m = waveM * 32 + ms * 16 + ln15;
#pragma unroll
            for (int v = 0; v < 8; ++v) {
                int k = (v < 4) ? (hf * 8 + 2 * v) : (16 + hf * 8 + 2 * (v - 4));
                fa[ms].u[v] = *(const unsigned*)(&as[m * GLSTR + k]);
            }
        }
#pragma unroll
        for (int ns = 0; ns < 4; ++ns) {
            int n = waveN * 64 + ns * 16 + ln15;
#pragma unroll
            for (int v = 0; v < 8; ++v) {
                int k = (v < 4) ? (hf * 8 + 2 * v) : (16 + hf * 8 + 2 * (v - 4));
                fb[ns].u[v] = *(const unsigned*)(&ws[n * GLSTR + k]);
            }
        }
#pragma unroll
        for (int ms = 0; ms < 2; ++ms)
#pragma unroll
            for (int ns = 0; ns < 4; ++ns)
                acc[ms][ns] = __builtin_amdgcn_wmma_f32_16x16x32_bf16(
                    false, fa[ms].v, false, fb[ns].v,
                    (short)0, acc[ms][ns], false, false);
    };

    const bool fullN = (bN + GBLK) <= N;   // uniform per block

    if (fullN) {
        // --- async double-buffered path -------------------------------------
        const int row = tid >> 2;          // 0..63
        const int col = (tid & 3) * 8;     // 0,8,16,24
        const __bf16* aP0 = A + (size_t)(bM + row) * K + col;
        const __bf16* aP1 = A + (size_t)(bM + row + 64) * K + col;
        const __bf16* wP0 = W + (size_t)(bN + row) * K + col;
        const __bf16* wP1 = W + (size_t)(bN + row + 64) * K + col;
        const int l0 = row * GLSTR + col;
        const int l1 = (row + 64) * GLSTR + col;

        auto stage = [&](int buf, int k0) {
            mamba_async_b128((unsigned)(uintptr_t)&As[buf][l0], aP0 + k0);
            mamba_async_b128((unsigned)(uintptr_t)&As[buf][l1], aP1 + k0);
            mamba_async_b128((unsigned)(uintptr_t)&Ws[buf][l0], wP0 + k0);
            mamba_async_b128((unsigned)(uintptr_t)&Ws[buf][l1], wP1 + k0);
        };

        stage(0, 0);
        mamba_wait_async0();
        __syncthreads();

        const int nk = K / GBK;
        for (int kt = 0; kt < nk; ++kt) {
            const int buf = kt & 1;
            if (kt + 1 < nk) stage(buf ^ 1, (kt + 1) * GBK);   // overlap with WMMA
            compute(As[buf], Ws[buf]);
            if (kt + 1 < nk) mamba_wait_async0();
            __syncthreads();
        }
    } else {
        // --- guarded synchronous fallback (ragged N, e.g. x_proj N=96) ------
        for (int k0 = 0; k0 < K; k0 += GBK) {
#pragma unroll
            for (int rep = 0; rep < 2; ++rep) {
                int idx = tid + rep * 256;
                int row = idx >> 2;
                int col = (idx & 3) * 8;
                const uint4* asrc =
                    (const uint4*)(A + (size_t)(bM + row) * K + k0 + col);
                *(uint4*)(&As[0][row * GLSTR + col]) = *asrc;
                int nrow = bN + row;
                uint4 wv = make_uint4(0u, 0u, 0u, 0u);
                if (nrow < N)
                    wv = *(const uint4*)(W + (size_t)nrow * K + k0 + col);
                *(uint4*)(&Ws[0][row * GLSTR + col]) = wv;
            }
            __syncthreads();
            compute(As[0], Ws[0]);
            __syncthreads();
        }
    }

    // Epilogue: C/D layout -> VGPR r holds M = r + hf*8, N = lane&15.
#pragma unroll
    for (int ms = 0; ms < 2; ++ms) {
#pragma unroll
        for (int ns = 0; ns < 4; ++ns) {
            int gn = bN + waveN * 64 + ns * 16 + ln15;
            if (gn >= N) continue;
#pragma unroll
            for (int r = 0; r < 8; ++r) {
                int gm = bM + waveM * 32 + ms * 16 + hf * 8 + r;
                C[(size_t)gm * N + gn] = acc[ms][ns][r];
            }
        }
    }
}

// --------------------- depthwise causal conv + SiLU --------------------------
__global__ void mamba_conv_silu_kernel(const float* __restrict__ xz,
                                       const float* __restrict__ cw,
                                       const float* __restrict__ cb,
                                       float* __restrict__ uc,
                                       __bf16* __restrict__ ub) {
    size_t idx = (size_t)blockIdx.x * 256 + threadIdx.x;
    const size_t total = (size_t)M_SZ * DI_SZ;
    if (idx >= total) return;
    const int d  = (int)(idx % DI_SZ);
    const size_t bt = idx / DI_SZ;              // b*L + t
    const int t  = (int)(bt % L_SZ);
    const size_t brow = bt - t;                 // b*L
    float acc = cb[d];
#pragma unroll
    for (int j = 0; j < 4; ++j) {
        int tt = t - 3 + j;
        if (tt >= 0)
            acc += xz[(brow + tt) * (size_t)(2 * DI_SZ) + d] * cw[d * 4 + j];
    }
    float s = acc / (1.f + __expf(-acc));       // SiLU
    uc[idx] = s;
    ub[idx] = (__bf16)s;
}

// ------------------------- dt bias + softplus --------------------------------
__global__ void mamba_softplus_kernel(float* __restrict__ dt,
                                      const float* __restrict__ dt_b) {
    size_t idx = (size_t)blockIdx.x * 256 + threadIdx.x;
    const size_t total = (size_t)M_SZ * DI_SZ;
    if (idx >= total) return;
    const int d = (int)(idx % DI_SZ);
    float v = dt[idx] + dt_b[d];
    dt[idx] = (v > 20.f) ? v : log1pf(__expf(v));
}

// ------------------------ selective scan + gate ------------------------------
__global__ __launch_bounds__(256)
void mamba_scan_kernel(const float* __restrict__ dt,
                       const float* __restrict__ dbc,
                       const float* __restrict__ uc,
                       const float* __restrict__ xz,
                       const float* __restrict__ A_log,
                       const float* __restrict__ Dp,
                       __bf16* __restrict__ yb) {
    const int d = blockIdx.x * 256 + threadIdx.x;   // channel
    const int b = blockIdx.y;                       // batch
    float a[DS_SZ], h[DS_SZ];
#pragma unroll
    for (int s = 0; s < DS_SZ; ++s) {
        a[s] = -__expf(A_log[(size_t)d * DS_SZ + s]);
        h[s] = 0.f;
    }
    const float Dv = Dp[d];
    __shared__ float sB[DS_SZ], sC[DS_SZ];
    const size_t rowBase = (size_t)b * L_SZ;

    for (int t = 0; t < L_SZ; ++t) {
        const size_t row = rowBase + t;
        if (threadIdx.x < 32) {
            float v = dbc[row * 96 + 64 + threadIdx.x];
            if (threadIdx.x < 16) sB[threadIdx.x] = v;
            else                  sC[threadIdx.x - 16] = v;
        }
        __syncthreads();
        const float dtv = dt[row * DI_SZ + d];
        const float uv  = uc[row * DI_SZ + d];
        const float zv  = xz[row * (size_t)(2 * DI_SZ) + DI_SZ + d];
        float y = 0.f;
#pragma unroll
        for (int s = 0; s < DS_SZ; ++s) {
            float dA = __expf(dtv * a[s]);
            h[s] = dA * h[s] + (dtv * sB[s]) * uv;
            y += h[s] * sC[s];
        }
        y += uv * Dv;
        y *= zv / (1.f + __expf(-zv));              // * silu(z)
        yb[row * DI_SZ + d] = (__bf16)y;
        __syncthreads();
    }
}

// ------------------------------ combine --------------------------------------
__global__ void mamba_combine_kernel(const float* __restrict__ x,
                                     const float* __restrict__ fo,
                                     const float* __restrict__ bo,
                                     float* __restrict__ out) {
    size_t idx = (size_t)blockIdx.x * 256 + threadIdx.x;
    const size_t total = (size_t)M_SZ * DM_SZ;
    if (idx >= total) return;
    const int m = (int)(idx % DM_SZ);
    const size_t bt = idx / DM_SZ;
    const int t = (int)(bt % L_SZ);
    const size_t b = bt / L_SZ;
    const size_t rev = ((size_t)b * L_SZ + (size_t)(L_SZ - 1 - t)) * DM_SZ + m;
    out[idx] = x[idx] + 0.5f * (fo[idx] + bo[rev]);
}

// ------------------------------ host side ------------------------------------
static inline size_t ws_align(size_t x) { return (x + 255) & ~(size_t)255; }

extern "C" void kernel_launch(void* const* d_in, const int* in_sizes, int n_in,
                              void* d_out, int out_size, void* d_ws, size_t ws_size,
                              hipStream_t stream) {
    (void)in_sizes; (void)n_in; (void)out_size; (void)ws_size;

    const float* x     = (const float*)d_in[0];
    const float* gamma = (const float*)d_in[1];
    const float* beta  = (const float*)d_in[2];

    char* ws = (char*)d_ws;
    size_t off = 0;
    auto alloc = [&](size_t bytes) -> void* {
        void* p = ws + off;
        off = ws_align(off + bytes);
        return p;
    };

    __bf16* lnF  = (__bf16*)alloc((size_t)M_SZ * DM_SZ * 2);
    __bf16* lnR  = (__bf16*)alloc((size_t)M_SZ * DM_SZ * 2);
    __bf16* wIn  = (__bf16*)alloc((size_t)2 * DI_SZ * DM_SZ * 2);
    __bf16* wXp  = (__bf16*)alloc((size_t)(DTR_SZ + 2 * DS_SZ) * DI_SZ * 2);
    __bf16* wDt  = (__bf16*)alloc((size_t)DI_SZ * DTR_SZ * 2);
    __bf16* wOut = (__bf16*)alloc((size_t)DM_SZ * DI_SZ * 2);
    float*  xz   = (float*) alloc((size_t)M_SZ * 2 * DI_SZ * 4);
    float*  ucv  = (float*) alloc((size_t)M_SZ * DI_SZ * 4);
    __bf16* ub   = (__bf16*)alloc((size_t)M_SZ * DI_SZ * 2);
    float*  dbc  = (float*) alloc((size_t)M_SZ * 96 * 4);
    __bf16* dtb  = (__bf16*)alloc((size_t)M_SZ * DTR_SZ * 2);
    float*  dtf  = (float*) alloc((size_t)M_SZ * DI_SZ * 4);
    __bf16* yb   = (__bf16*)alloc((size_t)M_SZ * DI_SZ * 2);
    float*  fOut = (float*) alloc((size_t)M_SZ * DM_SZ * 4);
    float*  bOut = (float*) alloc((size_t)M_SZ * DM_SZ * 4);

    // 1) LayerNorm -> bf16 activations (forward + time-reversed)
    mamba_layernorm_kernel<<<M_SZ, 256, 0, stream>>>(x, gamma, beta, lnF, lnR);

    for (int dir = 0; dir < 2; ++dir) {
        const int wb = dir ? 12 : 3;
        const float* in_w    = (const float*)d_in[wb + 0];
        const float* conv_w  = (const float*)d_in[wb + 1];
        const float* conv_b  = (const float*)d_in[wb + 2];
        const float* xproj_w = (const float*)d_in[wb + 3];
        const float* dt_w    = (const float*)d_in[wb + 4];
        const float* dt_b    = (const float*)d_in[wb + 5];
        const float* A_log   = (const float*)d_in[wb + 6];
        const float* Dvec    = (const float*)d_in[wb + 7];
        const float* out_w   = (const float*)d_in[wb + 8];

        // Weight casts to bf16 (reused buffers; directions run sequentially)
        {
            size_t n;
            n = (size_t)2 * DI_SZ * DM_SZ;
            mamba_cvt_bf16_kernel<<<(unsigned)((n + 255) / 256), 256, 0, stream>>>(in_w, wIn, n);
            n = (size_t)(DTR_SZ + 2 * DS_SZ) * DI_SZ;
            mamba_cvt_bf16_kernel<<<(unsigned)((n + 255) / 256), 256, 0, stream>>>(xproj_w, wXp, n);
            n = (size_t)DI_SZ * DTR_SZ;
            mamba_cvt_bf16_kernel<<<(unsigned)((n + 255) / 256), 256, 0, stream>>>(dt_w, wDt, n);
            n = (size_t)DM_SZ * DI_SZ;
            mamba_cvt_bf16_kernel<<<(unsigned)((n + 255) / 256), 256, 0, stream>>>(out_w, wOut, n);
        }

        const __bf16* lnA = dir ? lnR : lnF;

        // 2) in_proj: [4096,1024] x [4096,1024]^T -> xz [4096,4096]
        mamba_gemm_bf16_kernel<<<dim3(M_SZ / GBLK, (2 * DI_SZ + GBLK - 1) / GBLK),
                                 256, 0, stream>>>(lnA, wIn, xz, M_SZ, 2 * DI_SZ, DM_SZ);

        // 3) causal depthwise conv + bias + SiLU
        {
            size_t n = (size_t)M_SZ * DI_SZ;
            mamba_conv_silu_kernel<<<(unsigned)((n + 255) / 256), 256, 0, stream>>>(
                xz, conv_w, conv_b, ucv, ub);
        }

        // 4) x_proj: [4096,2048] x [96,2048]^T -> dbc [4096,96]
        mamba_gemm_bf16_kernel<<<dim3(M_SZ / GBLK, 1), 256, 0, stream>>>(
            ub, wXp, dbc, M_SZ, DTR_SZ + 2 * DS_SZ, DI_SZ);

        // 5) dt projection: slice dbc[:, :64] -> bf16, GEMM, bias+softplus
        {
            size_t n = (size_t)M_SZ * DTR_SZ;
            mamba_slice_bf16_kernel<<<(unsigned)((n + 255) / 256), 256, 0, stream>>>(
                dbc, dtb, 96, DTR_SZ, n);
        }
        mamba_gemm_bf16_kernel<<<dim3(M_SZ / GBLK, DI_SZ / GBLK), 256, 0, stream>>>(
            dtb, wDt, dtf, M_SZ, DI_SZ, DTR_SZ);
        {
            size_t n = (size_t)M_SZ * DI_SZ;
            mamba_softplus_kernel<<<(unsigned)((n + 255) / 256), 256, 0, stream>>>(dtf, dt_b);
        }

        // 6) selective scan + D-skip + SiLU(z) gating -> bf16 y
        mamba_scan_kernel<<<dim3(DI_SZ / 256, B_SZ), 256, 0, stream>>>(
            dtf, dbc, ucv, xz, A_log, Dvec, yb);

        // 7) out_proj: [4096,2048] x [1024,2048]^T -> per-direction output
        mamba_gemm_bf16_kernel<<<dim3(M_SZ / GBLK, DM_SZ / GBLK), 256, 0, stream>>>(
            yb, wOut, dir ? bOut : fOut, M_SZ, DM_SZ, DI_SZ);
    }

    // 8) out = x + 0.5*(f_out + flip(b_out))
    {
        size_t n = (size_t)M_SZ * DM_SZ;
        mamba_combine_kernel<<<(unsigned)((n + 255) / 256), 256, 0, stream>>>(
            x, fOut, bOut, (float*)d_out);
    }
}
